// CarlosSelfAttention_39101382263164
// MI455X (gfx1250) — matrix operand order repaired and checked by reading, not compiled, not run on hardware
//
#include <hip/hip_runtime.h>

// ---------------------------------------------------------------------------
// CDNA5 (gfx1250) wave32 WMMA self-attention.
// Shapes fixed by the reference: B=2, T=2048, C=1024, H=16, hd=64.
// Register-blocked: GEMMs 32x64/wave, attention 32 q-rows/wave.
// ---------------------------------------------------------------------------

typedef __attribute__((ext_vector_type(16))) __bf16 v16bf;
typedef __attribute__((ext_vector_type(8)))  float  v8f;

#define B_  2
#define T_  2048
#define C_  1024
#define H_  16
#define HD_ 64
#define BT_ (B_ * T_)
#define C3_ (3 * C_)

union FragU {
    v16bf v;
    uint4 q[2];
};

__device__ __forceinline__ v8f wmma_bf16(v16bf a, v16bf b, v8f c) {
    // D = A(16x32) * B(32x16) + C, f32 accumulate
    return __builtin_amdgcn_wmma_f32_16x16x32_bf16(
        /*neg_a=*/false, a, /*neg_b=*/false, b,
        /*c_mod=*/(short)0, c, /*reuse_a=*/false, /*reuse_b=*/false);
}

// A fragment (16x32 bf16): lane holds row (lane&15); per-lane data is two
// contiguous 8-elem chunks at K = half*8 and K = 16 + half*8 (ISA 7.12.2).
__device__ __forceinline__ v16bf load_frag_a(const __bf16* base, int row_stride,
                                             int lane, int k0) {
    FragU f;
    const __bf16* p = base + (size_t)(lane & 15) * row_stride;
    int c0 = k0 + ((lane >> 4) << 3);
    f.q[0] = *reinterpret_cast<const uint4*>(p + c0);
    f.q[1] = *reinterpret_cast<const uint4*>(p + c0 + 16);
    return f.v;
}

// B fragment (32x16 bf16) gathered from an [n][k] row-major source:
// lane holds column n = (lane&15); two 8-elem chunks at K = half*8 and K = 16 + half*8.
__device__ __forceinline__ v16bf load_frag_b(const __bf16* base, int row_stride,
                                             int lane, int k0) {
    FragU f;
    const __bf16* p = base + (size_t)(lane & 15) * row_stride;
    int c0 = k0 + ((lane >> 4) << 3);
    f.q[0] = *reinterpret_cast<const uint4*>(p + c0);
    f.q[1] = *reinterpret_cast<const uint4*>(p + c0 + 16);
    return f.v;
}

// ---------------------------------------------------------------------------
// Kernel 0: fp32 -> bf16 conversion (grid-stride)
// ---------------------------------------------------------------------------
__global__ void cvt_bf16_kernel(const float* __restrict__ in,
                                __bf16* __restrict__ out, int n) {
    for (int i = blockIdx.x * blockDim.x + threadIdx.x; i < n;
         i += gridDim.x * blockDim.x) {
        out[i] = (__bf16)in[i];
    }
}

// ---------------------------------------------------------------------------
// Kernel 1: qkv = x @ Wqkv^T + b, fused RoPE, scatter to Q/K (rope'd, Q also
// pre-scaled by 1/sqrt(hd)) in [b][h][t][d] and V transposed [b][h][d][t].
// One wave computes a 32x64 output tile (8 WMMAs per 32-wide K step).
// ---------------------------------------------------------------------------
__global__ __launch_bounds__(32)
void qkv_rope_kernel(const __bf16* __restrict__ xb,
                     const __bf16* __restrict__ wqkv,
                     const float* __restrict__ bqkv,
                     __bf16* __restrict__ Qr, __bf16* __restrict__ Kr,
                     __bf16* __restrict__ Vt) {
    const int lane = threadIdx.x;
    const int n0 = blockIdx.x * 32;   // token-tile (over B*T)
    const int o0 = blockIdx.y * 64;   // feature-tile (over 3C)
    const __bf16* arow0 = xb + (size_t)n0 * C_;
    const __bf16* arow1 = xb + (size_t)(n0 + 16) * C_;

    v8f acc[2][4] = {};
    for (int kk = 0; kk < C_; kk += 32) {
        __builtin_prefetch(arow0 + (size_t)(lane & 15) * C_ + kk + 128, 0, 3);
        v16bf a0 = load_frag_a(arow0, C_, lane, kk);
        v16bf a1 = load_frag_a(arow1, C_, lane, kk);
#pragma unroll
        for (int ct = 0; ct < 4; ++ct) {
            const __bf16* brow = wqkv + (size_t)(o0 + ct * 16) * C_;
            v16bf b = load_frag_b(brow, C_, lane, kk);
            acc[0][ct] = wmma_bf16(a0, b, acc[0][ct]);
            acc[1][ct] = wmma_bf16(a1, b, acc[1][ct]);
        }
    }

    const int half = lane >> 4;
    const int ncol = lane & 15;

#pragma unroll
    for (int ct = 0; ct < 4; ++ct) {
        const int o = o0 + ct * 16 + ncol;
        const int qkvt = o >> 10;            // 0=q 1=k 2=v
        const int within = o & (C_ - 1);
        const int h = within >> 6;
        const int d = within & (HD_ - 1);
        const float bias = bqkv[o];
        const int j = d & 31;                // inv_freq index
        const float ifreq = __powf(10000.0f, -(float)j * (1.0f / 32.0f));
#pragma unroll
        for (int rt = 0; rt < 2; ++rt) {
#pragma unroll
            for (int r = 0; r < 8; ++r) {
                const int token = n0 + rt * 16 + r + half * 8;
                const int bb = token >> 11;          // /T_
                const int t = token & (T_ - 1);
                float val = acc[rt][ct][r] + bias;
                // rotate_half partner = adjacent feature = adjacent lane
                float partner = __shfl_xor(val, 1, 32);
                if (qkvt == 2) {
                    // V transposed: [b][h][d][t]
                    Vt[(((size_t)bb * H_ + h) * HD_ + d) * T_ + t] = (__bf16)val;
                } else {
                    float ang = (float)t * ifreq;
                    float cs = __cosf(ang), sn = __sinf(ang);
                    float rot = (d & 1) ? partner : -partner;  // (-x_odd, x_even)
                    float out = val * cs + rot * sn;
                    size_t idx = (((size_t)bb * H_ + h) * T_ + t) * HD_ + d;
                    if (qkvt == 0)
                        Qr[idx] = (__bf16)(out * 0.125f);      // fold 1/sqrt(64)
                    else
                        Kr[idx] = (__bf16)out;
                }
            }
        }
    }
}

// ---------------------------------------------------------------------------
// Kernel 2: flash attention. One wave per (b,h,32-row q-tile).
// 32-key blocks: 8 S-WMMAs + 8 PV-WMMAs per block; K/V fragments shared
// across both 16-row halves of the q-tile. P staged through LDS for the
// C-layout -> A-layout transpose.
// ---------------------------------------------------------------------------
__global__ __launch_bounds__(32)
void attn_kernel(const __bf16* __restrict__ Qr, const __bf16* __restrict__ Kr,
                 const __bf16* __restrict__ Vt, __bf16* __restrict__ attnb) {
    __shared__ __bf16 P_lds[32 * 32];

    const int lane = threadIdx.x;
    const int qt = blockIdx.x & 63;     // T/32 = 64 q-tiles
    const int bh = blockIdx.x >> 6;     // 0..31
    const int q0 = qt * 32;
    const int half = lane >> 4;
    const int ncol = lane & 15;

    const __bf16* Qb = Qr + (size_t)bh * T_ * HD_;
    const __bf16* Kb = Kr + (size_t)bh * T_ * HD_;
    const __bf16* Vb = Vt + (size_t)bh * HD_ * T_;

    // Q tile: 2 row-tiles x 2 d-halves of 16x32 A fragments
    v16bf qa[2][2];
#pragma unroll
    for (int rt = 0; rt < 2; ++rt) {
        qa[rt][0] = load_frag_a(Qb + (size_t)(q0 + rt * 16) * HD_, HD_, lane, 0);
        qa[rt][1] = load_frag_a(Qb + (size_t)(q0 + rt * 16) * HD_, HD_, lane, 32);
    }

    v8f oa[2][4] = {};
    float m[2][8], l[2][8];
#pragma unroll
    for (int rt = 0; rt < 2; ++rt)
#pragma unroll
        for (int r = 0; r < 8; ++r) { m[rt][r] = -1e30f; l[rt][r] = 0.0f; }

    const int jmax = (q0 + 31) >> 5;    // last (diagonal) key block
    for (int j = 0; j <= jmax; ++j) {
        const int k0 = j * 32;

        // ---- S = Q K^T : K fragments shared across both q row-tiles ----
        v8f sc[2][2];                   // [row-tile][key col-tile]
#pragma unroll
        for (int nt = 0; nt < 2; ++nt) {
            const __bf16* kb = Kb + (size_t)(k0 + nt * 16) * HD_;
            v16bf b0 = load_frag_b(kb, HD_, lane, 0);
            v16bf b1 = load_frag_b(kb, HD_, lane, 32);
#pragma unroll
            for (int rt = 0; rt < 2; ++rt) {
                v8f s = {};
                s = wmma_bf16(qa[rt][0], b0, s);
                s = wmma_bf16(qa[rt][1], b1, s);
                sc[rt][nt] = s;
            }
        }

        // ---- causal mask (only the diagonal 32x32 block needs it) ----
        if (k0 + 31 > q0) {
#pragma unroll
            for (int rt = 0; rt < 2; ++rt)
#pragma unroll
                for (int nt = 0; nt < 2; ++nt)
#pragma unroll
                    for (int r = 0; r < 8; ++r) {
                        int qi = q0 + rt * 16 + r + half * 8;
                        int ki = k0 + nt * 16 + ncol;
                        if (ki > qi) sc[rt][nt][r] = -1e30f;
                    }
        }

        // ---- online softmax; write P (bf16) to LDS ----
#pragma unroll
        for (int rt = 0; rt < 2; ++rt) {
#pragma unroll
            for (int r = 0; r < 8; ++r) {
                float rm = fmaxf(sc[rt][0][r], sc[rt][1][r]);
                rm = fmaxf(rm, __shfl_xor(rm, 1, 32));
                rm = fmaxf(rm, __shfl_xor(rm, 2, 32));
                rm = fmaxf(rm, __shfl_xor(rm, 4, 32));
                rm = fmaxf(rm, __shfl_xor(rm, 8, 32)); // within 16-lane half
                float mn = fmaxf(m[rt][r], rm);
                float scale = __expf(m[rt][r] - mn);
                float p0 = __expf(sc[rt][0][r] - mn);
                float p1 = __expf(sc[rt][1][r] - mn);
                float rs = p0 + p1;
                rs += __shfl_xor(rs, 1, 32);
                rs += __shfl_xor(rs, 2, 32);
                rs += __shfl_xor(rs, 4, 32);
                rs += __shfl_xor(rs, 8, 32);
                l[rt][r] = l[rt][r] * scale + rs;
                m[rt][r] = mn;
                const int M = rt * 16 + r + half * 8;
                P_lds[M * 32 + ncol] = (__bf16)p0;
                P_lds[M * 32 + 16 + ncol] = (__bf16)p1;
                oa[rt][0][r] *= scale; oa[rt][1][r] *= scale;
                oa[rt][2][r] *= scale; oa[rt][3][r] *= scale;
            }
        }

        asm volatile("s_wait_dscnt 0" ::: "memory");

        // ---- reload P in A-fragment layout (two 16x32 fragments) ----
        FragU pf[2];
#pragma unroll
        for (int rt = 0; rt < 2; ++rt) {
            const int row = rt * 16 + (lane & 15);
            pf[rt].q[0] = *reinterpret_cast<const uint4*>(
                &P_lds[row * 32 + ((lane >> 4) << 3)]);
            pf[rt].q[1] = *reinterpret_cast<const uint4*>(
                &P_lds[row * 32 + 16 + ((lane >> 4) << 3)]);
        }

        // ---- O += P @ V : each V fragment shared by both row-tiles ----
#pragma unroll
        for (int t = 0; t < 4; ++t) {
            v16bf vb = load_frag_b(Vb + (size_t)(t * 16) * T_, T_, lane, k0);
            oa[0][t] = wmma_bf16(pf[0].v, vb, oa[0][t]);
            oa[1][t] = wmma_bf16(pf[1].v, vb, oa[1][t]);
        }
    }

    // ---- epilogue: O / l -> attn output [b][t][h*64 + d] (bf16) ----
    const int bb = bh >> 4;
    const int h = bh & 15;
#pragma unroll
    for (int rt = 0; rt < 2; ++rt) {
#pragma unroll
        for (int r = 0; r < 8; ++r) {
            const int tok = q0 + rt * 16 + r + half * 8;
            const size_t rowbase = ((size_t)bb * T_ + tok) * C_ + h * HD_;
            const float inv_l = 1.0f / l[rt][r];
            attnb[rowbase + 0 + ncol]  = (__bf16)(oa[rt][0][r] * inv_l);
            attnb[rowbase + 16 + ncol] = (__bf16)(oa[rt][1][r] * inv_l);
            attnb[rowbase + 32 + ncol] = (__bf16)(oa[rt][2][r] * inv_l);
            attnb[rowbase + 48 + ncol] = (__bf16)(oa[rt][3][r] * inv_l);
        }
    }
}

// ---------------------------------------------------------------------------
// Kernel 3: out = attn @ Wo^T + bo (fp32 output), 32x64 tile per wave
// ---------------------------------------------------------------------------
__global__ __launch_bounds__(32)
void oproj_kernel(const __bf16* __restrict__ attnb,
                  const __bf16* __restrict__ wob,
                  const float* __restrict__ bo, float* __restrict__ out) {
    const int lane = threadIdx.x;
    const int n0 = blockIdx.x * 32;
    const int o0 = blockIdx.y * 64;
    const __bf16* arow0 = attnb + (size_t)n0 * C_;
    const __bf16* arow1 = attnb + (size_t)(n0 + 16) * C_;

    v8f acc[2][4] = {};
    for (int kk = 0; kk < C_; kk += 32) {
        __builtin_prefetch(arow0 + (size_t)(lane & 15) * C_ + kk + 128, 0, 3);
        v16bf a0 = load_frag_a(arow0, C_, lane, kk);
        v16bf a1 = load_frag_a(arow1, C_, lane, kk);
#pragma unroll
        for (int ct = 0; ct < 4; ++ct) {
            const __bf16* brow = wob + (size_t)(o0 + ct * 16) * C_;
            v16bf b = load_frag_b(brow, C_, lane, kk);
            acc[0][ct] = wmma_bf16(a0, b, acc[0][ct]);
            acc[1][ct] = wmma_bf16(a1, b, acc[1][ct]);
        }
    }

    const int half = lane >> 4;
    const int ncol = lane & 15;
#pragma unroll
    for (int ct = 0; ct < 4; ++ct) {
        const float bias = bo[o0 + ct * 16 + ncol];
#pragma unroll
        for (int rt = 0; rt < 2; ++rt) {
#pragma unroll
            for (int r = 0; r < 8; ++r) {
                const int row = n0 + rt * 16 + r + half * 8;
                out[(size_t)row * C_ + o0 + ct * 16 + ncol] =
                    acc[rt][ct][r] + bias;
            }
        }
    }
}

// ---------------------------------------------------------------------------
// Host-side launcher
// ---------------------------------------------------------------------------
extern "C" void kernel_launch(void* const* d_in, const int* in_sizes, int n_in,
                              void* d_out, int out_size, void* d_ws, size_t ws_size,
                              hipStream_t stream) {
    const float* x    = (const float*)d_in[0];
    // d_in[1] = mask (causal; handled analytically)
    const float* Wqkv = (const float*)d_in[2];
    const float* bqkv = (const float*)d_in[3];
    const float* Wo   = (const float*)d_in[4];
    const float* bo   = (const float*)d_in[5];
    float* out = (float*)d_out;

    // workspace carve-up (all bf16), 48 MB total
    char* ws = (char*)d_ws;
    const size_t SZ_X    = (size_t)BT_ * C_ * 2;            // 8 MB
    const size_t SZ_WQKV = (size_t)C3_ * C_ * 2;            // 6 MB
    const size_t SZ_WO   = (size_t)C_ * C_ * 2;             // 2 MB
    const size_t SZ_HEAD = (size_t)B_ * H_ * T_ * HD_ * 2;  // 8 MB each

    __bf16* xb    = (__bf16*)(ws);
    __bf16* wqkvb = (__bf16*)(ws + SZ_X);
    __bf16* wob   = (__bf16*)(ws + SZ_X + SZ_WQKV);
    __bf16* Qr    = (__bf16*)(ws + SZ_X + SZ_WQKV + SZ_WO);
    __bf16* Kr    = (__bf16*)(ws + SZ_X + SZ_WQKV + SZ_WO + SZ_HEAD);
    __bf16* Vt    = (__bf16*)(ws + SZ_X + SZ_WQKV + SZ_WO + 2 * SZ_HEAD);
    __bf16* attnb = (__bf16*)(ws + SZ_X + SZ_WQKV + SZ_WO + 3 * SZ_HEAD);

    // 0) fp32 -> bf16
    cvt_bf16_kernel<<<4096, 256, 0, stream>>>(x, xb, BT_ * C_);
    cvt_bf16_kernel<<<4096, 256, 0, stream>>>(Wqkv, wqkvb, C3_ * C_);
    cvt_bf16_kernel<<<2048, 256, 0, stream>>>(Wo, wob, C_ * C_);

    // 1) QKV projection + RoPE + scatter (32x64 tiles)
    qkv_rope_kernel<<<dim3(BT_ / 32, C3_ / 64), 32, 0, stream>>>(
        xb, wqkvb, bqkv, Qr, Kr, Vt);

    // 2) flash attention: B*H * (T/32) waves
    attn_kernel<<<B_ * H_ * (T_ / 32), 32, 0, stream>>>(Qr, Kr, Vt, attnb);

    // 3) output projection (32x64 tiles)
    oproj_kernel<<<dim3(BT_ / 32, C_ / 64), 32, 0, stream>>>(attnb, wob, bo, out);
}
